// PVSemanticGuidance_17540646437672
// MI455X (gfx1250) — matrix operand order repaired and checked by reading, not compiled, not run on hardware
//
#include <hip/hip_runtime.h>

#define C_    128
#define HEADS 4
#define HD    32
#define B_    2
#define HW    4096

typedef _Float16 half16_t __attribute__((ext_vector_type(16)));
typedef __fp16   fp16x2_t __attribute__((ext_vector_type(2)));
typedef float    float8_t __attribute__((ext_vector_type(8)));

union Frag {
    half16_t h;
    uint4    q[2];
    unsigned w[8];
};

// single v_cvt_pk_rtz_f16_f32
__device__ inline unsigned pk2(float a, float b) {
    union { fp16x2_t h; unsigned u; } x;
    x.h = __builtin_amdgcn_cvt_pkrtz(a, b);
    return x.u;
}

// raw v_exp_f32 (flushes to 0 below ~-126, fine for softmax)
__device__ inline float fexp2(float x) { return __builtin_amdgcn_exp2f(x); }

// wave32 xor-16 half swap in one VALU op: v_permlanex16_b32 with identity selects
__device__ inline int xor16i(int v) {
    return __builtin_amdgcn_permlanex16(v, v, 0x76543210, 0xfedcba98, false, false);
}
__device__ inline float xor16f(float v) {
    return __int_as_float(xor16i(__float_as_int(v)));
}

// ---------------------------------------------------------------------------
// mask_embed conv1: (B,1,64,64) -> relu(conv3x3) -> hidden (B,32,64,64)
// ---------------------------------------------------------------------------
__global__ __launch_bounds__(256) void mask1_kernel(
    const float* __restrict__ seg, const float* __restrict__ Wm1,
    const float* __restrict__ bm1, float* __restrict__ hidden) {
    int t  = blockIdx.x * 256 + threadIdx.x;      // over B*HW
    int b  = t >> 12;
    int hw = t & 4095;
    int y = hw >> 6, x = hw & 63;
    float in[9];
#pragma unroll
    for (int dy = -1; dy <= 1; ++dy)
#pragma unroll
        for (int dx = -1; dx <= 1; ++dx) {
            int yy = y + dy, xx = x + dx;
            bool ok = (yy >= 0 && yy < 64 && xx >= 0 && xx < 64);
            in[(dy + 1) * 3 + dx + 1] = ok ? seg[(size_t)b * HW + yy * 64 + xx] : 0.f;
        }
    for (int oc = 0; oc < 32; ++oc) {
        float acc = bm1[oc];
#pragma unroll
        for (int k = 0; k < 9; ++k) acc += Wm1[oc * 9 + k] * in[k];
        hidden[((size_t)(b * 32 + oc)) * HW + hw] = fmaxf(acc, 0.f);
    }
}

// ---------------------------------------------------------------------------
// mask_embed conv2: hidden (B,32,64,64) -> mask_feat (B,128,64,64)
// grid: (HW/256, 128, B)
// ---------------------------------------------------------------------------
__global__ __launch_bounds__(256) void mask2_kernel(
    const float* __restrict__ hidden, const float* __restrict__ Wm2,
    const float* __restrict__ bm2, float* __restrict__ mf) {
    int hw = blockIdx.x * 256 + threadIdx.x;
    int oc = blockIdx.y, b = blockIdx.z;
    int y = hw >> 6, x = hw & 63;
    float acc = bm2[oc];
    for (int ic = 0; ic < 32; ++ic) {
        const float* hp = hidden + ((size_t)(b * 32 + ic)) * HW;
        const float* wp = Wm2 + (oc * 32 + ic) * 9;
#pragma unroll
        for (int dy = -1; dy <= 1; ++dy) {
            int yy = y + dy;
            if (yy < 0 || yy > 63) continue;
#pragma unroll
            for (int dx = -1; dx <= 1; ++dx) {
                int xx = x + dx;
                if (xx < 0 || xx > 63) continue;
                acc += wp[(dy + 1) * 3 + (dx + 1)] * hp[yy * 64 + xx];
            }
        }
    }
    mf[((size_t)(b * C_) + oc) * HW + hw] = acc;
}

// ---------------------------------------------------------------------------
// QKV 1x1 projections + f16 packing.
//   Qh[b][h][row][d]  (f16, pre-scaled by SCALE*log2e)
//   Kh[b][h][key][d]  (f16)
//   Vt[b][h][d][key]  (f16, transposed)
// grid: (HW/256, 128, B)
// ---------------------------------------------------------------------------
__global__ __launch_bounds__(256) void qkv_kernel(
    const float* __restrict__ sr, const float* __restrict__ mf,
    const float* __restrict__ Wq, const float* __restrict__ bq,
    const float* __restrict__ Wk, const float* __restrict__ bk,
    const float* __restrict__ Wv, const float* __restrict__ bv,
    _Float16* __restrict__ Qh, _Float16* __restrict__ Kh,
    _Float16* __restrict__ Vt) {
    const int hw = blockIdx.x * 256 + threadIdx.x;
    const int c = blockIdx.y;
    const int b = blockIdx.z;
    const float* sp = sr + ((size_t)b * C_) * HW + hw;
    const float* mp = mf + ((size_t)b * C_) * HW + hw;
    float aq = 0.f, ak = 0.f, av = 0.f;
    for (int ci = 0; ci < C_; ++ci) {
        float s = sp[(size_t)ci * HW];
        float m = mp[(size_t)ci * HW];
        aq += Wq[c * C_ + ci] * s;
        ak += Wk[c * C_ + ci] * m;
        av += Wv[c * C_ + ci] * m;
    }
    aq += bq[c]; ak += bk[c]; av += bv[c];
    const int h = c >> 5, d = c & 31;
    const float QSCALE = 0.17677669529663687f * 1.4426950408889634f; // 32^-0.5 * log2(e)
    Qh[(((size_t)(b * HEADS + h)) * HW + hw) * HD + d] = (_Float16)(aq * QSCALE);
    Kh[(((size_t)(b * HEADS + h)) * HW + hw) * HD + d] = (_Float16)ak;
    Vt[(((size_t)(b * HEADS + h)) * HD + d) * HW + hw] = (_Float16)av;
}

// ---------------------------------------------------------------------------
// Flash attention, fully in registers. One wave32 per 16-query tile.
// S^T = K * Q^T  (lane index = query row -> 1-op softmax reductions)
// D^T = V^T * P^T (P^T fragment built via 8 packed v_permlanex16 exchanges)
// ---------------------------------------------------------------------------
__global__ __launch_bounds__(256) void attn_kernel(
    const _Float16* __restrict__ Qh, const _Float16* __restrict__ Kh,
    const _Float16* __restrict__ Vt, float* __restrict__ guided) {
    const int lane = threadIdx.x & 31;
    const int wid  = blockIdx.x * (blockDim.x >> 5) + (threadIdx.x >> 5);
    const int qt = wid & 255;          // 256 query tiles of 16 rows
    const int h  = (wid >> 8) & 3;
    const int b  = wid >> 10;
    const int qbase = qt * 16;
    const int lh  = lane & 15;
    const int sel = lane >> 4;

    const _Float16* Qp = Qh + (size_t)(b * HEADS + h) * HW * HD;
    const _Float16* Kp = Kh + (size_t)(b * HEADS + h) * HW * HD;
    const _Float16* Vp = Vt + (size_t)(b * HEADS + h) * HD * HW;

    // Q^T B-fragment (loop invariant): lane<16 -> Q[qbase+lh][0..7,16..23], else [8..15,24..31]
    Frag qf;
    {
        const _Float16* p = Qp + (size_t)(qbase + lh) * HD + sel * 8;
        qf.q[0] = ((const uint4*)p)[0];
        qf.q[1] = ((const uint4*)(p + 16))[0];
    }

    float8_t Oa = {}; // D^T dims 0..15
    float8_t Ob = {}; // D^T dims 16..31
    float m_state = -3.0e38f;
    float l_state = 0.0f;

    for (int kb = 0; kb < HW; kb += 32) {
        // K A-fragments (16 keys x 32 dims), two tiles
        Frag ka, kbf, va, vb;
        {
            const _Float16* p0 = Kp + (size_t)(kb + lh) * HD + sel * 8;
            ka.q[0]  = ((const uint4*)p0)[0];
            ka.q[1]  = ((const uint4*)(p0 + 16))[0];
            const _Float16* p1 = Kp + (size_t)(kb + 16 + lh) * HD + sel * 8;
            kbf.q[0] = ((const uint4*)p1)[0];
            kbf.q[1] = ((const uint4*)(p1 + 16))[0];
            // V^T A-fragments (16 dims x 32 keys), dims 0..15 and 16..31
            const _Float16* pv0 = Vp + (size_t)lh * HW + kb + sel * 16;
            va.q[0]  = ((const uint4*)pv0)[0];
            va.q[1]  = ((const uint4*)pv0)[1];
            const _Float16* pv1 = Vp + (size_t)(16 + lh) * HW + kb + sel * 16;
            vb.q[0]  = ((const uint4*)pv1)[0];
            vb.q[1]  = ((const uint4*)pv1)[1];
        }

        float8_t z = {};
        // S^T tiles: [key][qrow]; lane = qrow, vgpr r = key (r or 8+r by half)
        float8_t Sa = __builtin_amdgcn_wmma_f32_16x16x32_f16(false, ka.h,  false, qf.h, (short)0, z, false, false);
        float8_t Sb = __builtin_amdgcn_wmma_f32_16x16x32_f16(false, kbf.h, false, qf.h, (short)0, z, false, false);

        // online softmax in exp2 domain (SCALE*log2e folded into Q)
        float cm = Sa[0];
#pragma unroll
        for (int r = 0; r < 8; ++r) { cm = fmaxf(cm, Sa[r]); cm = fmaxf(cm, Sb[r]); }
        cm = fmaxf(cm, xor16f(cm));
        float m_new = fmaxf(m_state, cm);
        float alpha = fexp2(m_state - m_new);
        m_state = m_new;

        float sum = 0.f;
#pragma unroll
        for (int r = 0; r < 8; ++r) {
            Sa[r] = fexp2(Sa[r] - m_new);
            Sb[r] = fexp2(Sb[r] - m_new);
            sum += Sa[r] + Sb[r];
        }
        sum += xor16f(sum);
        l_state = l_state * alpha + sum;

        // Build P^T B-fragment (32 keys x 16 qrows) via cross-half exchange
        unsigned pa[4], pb[4];
#pragma unroll
        for (int j = 0; j < 4; ++j) {
            pa[j] = pk2(Sa[2 * j], Sa[2 * j + 1]);
            pb[j] = pk2(Sb[2 * j], Sb[2 * j + 1]);
        }
        Frag pf;
#pragma unroll
        for (int j = 0; j < 4; ++j) {
            unsigned xa = (unsigned)xor16i((int)pa[j]);
            unsigned xb = (unsigned)xor16i((int)pb[j]);
            pf.w[j]     = sel ? xb    : pa[j];
            pf.w[j + 4] = sel ? pb[j] : xa;
        }

#pragma unroll
        for (int r = 0; r < 8; ++r) { Oa[r] *= alpha; Ob[r] *= alpha; }
        Oa = __builtin_amdgcn_wmma_f32_16x16x32_f16(false, va.h, false, pf.h, (short)0, Oa, false, false);
        Ob = __builtin_amdgcn_wmma_f32_16x16x32_f16(false, vb.h, false, pf.h, (short)0, Ob, false, false);
    }

    const float inv_l = 1.0f / l_state;
    const int qrow = qbase + lh;
    float* gp = guided + (size_t)b * C_ * HW + (size_t)h * HD * HW + qrow;
#pragma unroll
    for (int r = 0; r < 8; ++r) {
        int d0 = r + sel * 8;
        gp[(size_t)d0 * HW]        = Oa[r] * inv_l;
        gp[(size_t)(d0 + 16) * HW] = Ob[r] * inv_l;
    }
}

// ---------------------------------------------------------------------------
// Output 1x1 projection + residual. grid: (HW/256, 128, B)
// ---------------------------------------------------------------------------
__global__ __launch_bounds__(256) void outproj_kernel(
    const float* __restrict__ guided, const float* __restrict__ Wo,
    const float* __restrict__ bo, const float* __restrict__ sr,
    float* __restrict__ out) {
    int hw = blockIdx.x * 256 + threadIdx.x;
    int c = blockIdx.y, b = blockIdx.z;
    const float* gp = guided + ((size_t)b * C_) * HW + hw;
    float acc = bo[c];
    for (int ci = 0; ci < C_; ++ci)
        acc += Wo[c * C_ + ci] * gp[(size_t)ci * HW];
    size_t o = ((size_t)(b * C_) + c) * HW + hw;
    out[o] = acc + sr[o];
}

extern "C" void kernel_launch(void* const* d_in, const int* in_sizes, int n_in,
                              void* d_out, int out_size, void* d_ws, size_t ws_size,
                              hipStream_t stream) {
    const float* sr  = (const float*)d_in[0];
    const float* seg = (const float*)d_in[1];
    const float* Wq  = (const float*)d_in[2];
    const float* bq  = (const float*)d_in[3];
    const float* Wm1 = (const float*)d_in[4];
    const float* bm1 = (const float*)d_in[5];
    const float* Wm2 = (const float*)d_in[6];
    const float* bm2 = (const float*)d_in[7];
    const float* Wk  = (const float*)d_in[8];
    const float* bk  = (const float*)d_in[9];
    const float* Wv  = (const float*)d_in[10];
    const float* bv  = (const float*)d_in[11];
    const float* Wo  = (const float*)d_in[12];
    const float* bo  = (const float*)d_in[13];
    float* out = (float*)d_out;

    char* w = (char*)d_ws;
    float*    hidden = (float*)(w);                      // 2*32*4096*4  = 1 MB
    float*    mfeat  = (float*)(w +  1 * (1u << 20));    // 2*128*4096*4 = 4 MB
    _Float16* Qh     = (_Float16*)(w +  5 * (1u << 20)); // 2 MB
    _Float16* Kh     = (_Float16*)(w +  7 * (1u << 20)); // 2 MB
    _Float16* Vt     = (_Float16*)(w +  9 * (1u << 20)); // 2 MB
    float*    guided = (float*)(w + 11 * (1u << 20));    // 4 MB

    mask1_kernel<<<(B_ * HW) / 256, 256, 0, stream>>>(seg, Wm1, bm1, hidden);
    mask2_kernel<<<dim3(HW / 256, C_, B_), 256, 0, stream>>>(hidden, Wm2, bm2, mfeat);
    qkv_kernel<<<dim3(HW / 256, C_, B_), 256, 0, stream>>>(sr, mfeat, Wq, bq, Wk, bk,
                                                           Wv, bv, Qh, Kh, Vt);
    attn_kernel<<<256, 256, 0, stream>>>(Qh, Kh, Vt, guided); // 2048 wave tasks
    outproj_kernel<<<dim3(HW / 256, C_, B_), 256, 0, stream>>>(guided, Wo, bo, sr, out);
}
